// Seq2Seq_3178275799402
// MI455X (gfx1250) — compile-verified, matrix operand order checked by reading, not run on hardware
//
#include <hip/hip_runtime.h>
#include <hip/hip_bf16.h>
#include <cstddef>

// ---------------------------------------------------------------------------
// Seq2Seq LSTM encoder/decoder for MI455X (gfx1250, wave32, WMMA).
//
// Sizes (fixed): V=32000, E=512, H=1024, B=64, S=128, T=64.
// All GEMMs run as v_wmma_f32_16x16x32_bf16; weights converted per call to
// bf16 in workspace (~90MB total -> resident in the 192MB L2 across the
// 191 sequential timesteps). K-loops are software-pipelined so loads for
// step k+32 are in flight while the WMMAs for step k execute (avoids the
// s_wait_loadcnt 0 before every WMMA seen in the unpipelined version).
// ---------------------------------------------------------------------------

typedef __bf16 v16bf __attribute__((ext_vector_type(16)));
typedef float  v8f   __attribute__((ext_vector_type(8)));

#define WMMA_BF16(A, B, C) \
    __builtin_amdgcn_wmma_f32_16x16x32_bf16(false, (A), false, (B), (short)0, (C), false, false)

static __device__ __forceinline__ unsigned short f2bf(float f) {
    unsigned int u = __float_as_uint(f);
    unsigned int r = u + 0x7FFFu + ((u >> 16) & 1u);   // round-to-nearest-even
    return (unsigned short)(r >> 16);
}

static __device__ __forceinline__ float sigf(float x) {
    return 1.0f / (1.0f + __expf(-x));
}

// Load one 16x32 bf16 fragment (A-layout per ISA 7.12.2; B of g = x*W^T uses
// the identical pattern on the row-major NxK weight matrix).
//   lane 0..15  : row = lane,    elems 0..7 = K[k..k+7],    8..15 = K[k+16..k+23]
//   lane 16..31 : row = lane-16, elems 0..7 = K[k+8..k+15], 8..15 = K[k+24..k+31]
static __device__ __forceinline__ v16bf load_frag(const unsigned short* __restrict__ base,
                                                  int row0, int ld, int k) {
    const int lane = threadIdx.x & 31;
    const int r    = lane & 15;
    const int hh   = (lane & 16) ? 8 : 0;
    const unsigned short* p = base + (size_t)(row0 + r) * (size_t)ld + (size_t)(k + hh);
    union { uint4 q[2]; v16bf v; } f;
    f.q[0] = *(const uint4*)(p);        // 8 bf16 (16B, aligned)
    f.q[1] = *(const uint4*)(p + 16);   // 8 bf16 at K+16
    return f.v;
}

// ---------------------------------------------------------------------------
// One LSTM timestep:  g = x(64xE) * Wih^T + h(64xH) * Whh^T + b ; gate fuse.
// 512 waves: 256 tiles (4 m-tiles x 64 j-tiles) x 2 K-splits. Each block owns
// 4 tiles x 2 K-halves; the k-half-1 wave dumps its 4 accumulators to LDS and
// the k-half-0 wave reduces + applies the gate nonlinearities.
// A-fragment is reused across the 4 gate accumulators.
// ---------------------------------------------------------------------------
__global__ void __launch_bounds__(256) lstm_step_kernel(
    const unsigned short* __restrict__ x,     // 64 x 512  bf16
    const unsigned short* __restrict__ h,     // 64 x 1024 bf16
    const unsigned short* __restrict__ Wih,   // 4096 x 512  bf16 (row n, col k)
    const unsigned short* __restrict__ Whh,   // 4096 x 1024 bf16
    const float* __restrict__ bias,           // 4096 f32
    float* __restrict__ cst,                  // 64 x 1024 f32 (in-place)
    unsigned short* __restrict__ h_out)       // 64 x 1024 bf16
{
    const int widb  = threadIdx.x >> 5;          // 0..7 wave-in-block
    const int tileL = widb & 3;                  // tile slot in block
    const int kh    = widb >> 2;                 // K-split half: 0 or 1
    const int tile  = blockIdx.x * 4 + tileL;    // 0..255
    const int m0    = (tile >> 6) << 4;          // 0,16,32,48
    const int j0    = (tile & 63) << 4;          // 0..1008

    v8f acc[4] = {{}, {}, {}, {}};

    // ---- input->hidden: this wave covers K in [kh*256, kh*256+256) ----
    {
        int k = kh * 256;
        const int kend = k + 256;
        v16bf a  = load_frag(x, m0, 512, k);
        v16bf b0 = load_frag(Wih, j0,        512, k);
        v16bf b1 = load_frag(Wih, 1024 + j0, 512, k);
        v16bf b2 = load_frag(Wih, 2048 + j0, 512, k);
        v16bf b3 = load_frag(Wih, 3072 + j0, 512, k);
        for (; k + 32 < kend; k += 32) {
            v16bf an = load_frag(x, m0, 512, k + 32);
            v16bf c0 = load_frag(Wih, j0,        512, k + 32);
            v16bf c1 = load_frag(Wih, 1024 + j0, 512, k + 32);
            v16bf c2 = load_frag(Wih, 2048 + j0, 512, k + 32);
            v16bf c3 = load_frag(Wih, 3072 + j0, 512, k + 32);
            acc[0] = WMMA_BF16(a, b0, acc[0]);
            acc[1] = WMMA_BF16(a, b1, acc[1]);
            acc[2] = WMMA_BF16(a, b2, acc[2]);
            acc[3] = WMMA_BF16(a, b3, acc[3]);
            a = an; b0 = c0; b1 = c1; b2 = c2; b3 = c3;
        }
        acc[0] = WMMA_BF16(a, b0, acc[0]);
        acc[1] = WMMA_BF16(a, b1, acc[1]);
        acc[2] = WMMA_BF16(a, b2, acc[2]);
        acc[3] = WMMA_BF16(a, b3, acc[3]);
    }

    // ---- hidden->hidden: this wave covers K in [kh*512, kh*512+512) ----
    {
        int k = kh * 512;
        const int kend = k + 512;
        v16bf a  = load_frag(h, m0, 1024, k);
        v16bf b0 = load_frag(Whh, j0,        1024, k);
        v16bf b1 = load_frag(Whh, 1024 + j0, 1024, k);
        v16bf b2 = load_frag(Whh, 2048 + j0, 1024, k);
        v16bf b3 = load_frag(Whh, 3072 + j0, 1024, k);
        for (; k + 32 < kend; k += 32) {
            v16bf an = load_frag(h, m0, 1024, k + 32);
            v16bf c0 = load_frag(Whh, j0,        1024, k + 32);
            v16bf c1 = load_frag(Whh, 1024 + j0, 1024, k + 32);
            v16bf c2 = load_frag(Whh, 2048 + j0, 1024, k + 32);
            v16bf c3 = load_frag(Whh, 3072 + j0, 1024, k + 32);
            acc[0] = WMMA_BF16(a, b0, acc[0]);
            acc[1] = WMMA_BF16(a, b1, acc[1]);
            acc[2] = WMMA_BF16(a, b2, acc[2]);
            acc[3] = WMMA_BF16(a, b3, acc[3]);
            a = an; b0 = c0; b1 = c1; b2 = c2; b3 = c3;
        }
        acc[0] = WMMA_BF16(a, b0, acc[0]);
        acc[1] = WMMA_BF16(a, b1, acc[1]);
        acc[2] = WMMA_BF16(a, b2, acc[2]);
        acc[3] = WMMA_BF16(a, b3, acc[3]);
    }

    // ---- cross-K-half reduction in LDS (conflict-free: r*32+lane) ----
    __shared__ float red[4][4][256];
    const int lane = threadIdx.x & 31;
    if (kh == 1) {
        #pragma unroll
        for (int g = 0; g < 4; ++g)
            #pragma unroll
            for (int r = 0; r < 8; ++r)
                red[tileL][g][r * 32 + lane] = acc[g][r];
    }
    __syncthreads();
    if (kh == 0) {
        #pragma unroll
        for (int g = 0; g < 4; ++g)
            #pragma unroll
            for (int r = 0; r < 8; ++r)
                acc[g][r] += red[tileL][g][r * 32 + lane];

        // Epilogue. D layout: lane L holds column n = j0+(L&15), rows mb..mb+7.
        const int n  = j0 + (lane & 15);
        const int mb = m0 + ((lane & 16) ? 8 : 0);
        const float bi = bias[n];
        const float bf = bias[1024 + n];
        const float bg = bias[2048 + n];
        const float bo = bias[3072 + n];
        #pragma unroll
        for (int r = 0; r < 8; ++r) {
            const int idx = (mb + r) * 1024 + n;
            const float ig = sigf(acc[0][r] + bi);
            const float fg = sigf(acc[1][r] + bf);
            const float gg = tanhf(acc[2][r] + bg);
            const float og = sigf(acc[3][r] + bo);
            const float cn = fg * cst[idx] + ig * gg;
            cst[idx]   = cn;
            h_out[idx] = f2bf(og * tanhf(cn));
        }
    }
}

// ---------------------------------------------------------------------------
// logits[:, t, :] = h(64x1024) * Wfc^T(32000x1024) + bfc   (f32 out)
// 8000 waves = 2000 N-tiles x 4 M-tiles; K-loop software-pipelined.
// ---------------------------------------------------------------------------
__global__ void __launch_bounds__(256) logits_kernel(
    const unsigned short* __restrict__ h,     // 64 x 1024 bf16
    const unsigned short* __restrict__ Wfc,   // 32000 x 1024 bf16
    const float* __restrict__ bfc,            // 32000
    float* __restrict__ out,                  // (64, 64, 32000) f32
    int t)
{
    const int wid = (blockIdx.x << 3) | (threadIdx.x >> 5);   // 0..7999
    const int n0  = (wid % 2000) << 4;
    const int m0  = (wid / 2000) << 4;

    v8f acc = {};
    int k = 0;
    v16bf a = load_frag(h,   m0, 1024, 0);
    v16bf b = load_frag(Wfc, n0, 1024, 0);
    for (; k + 32 < 1024; k += 32) {
        v16bf an = load_frag(h,   m0, 1024, k + 32);
        v16bf bn = load_frag(Wfc, n0, 1024, k + 32);
        acc = WMMA_BF16(a, b, acc);
        a = an; b = bn;
    }
    acc = WMMA_BF16(a, b, acc);

    const int lane = threadIdx.x & 31;
    const int n    = n0 + (lane & 15);
    const int mb   = m0 + ((lane & 16) ? 8 : 0);
    const float bv = bfc[n];
    #pragma unroll
    for (int r = 0; r < 8; ++r) {
        const size_t m = (size_t)(mb + r);
        out[m * (64u * 32000u) + (size_t)t * 32000u + (size_t)n] = acc[r] + bv;
    }
}

// Per-batch-row argmax (first-index tie-break) + teacher forcing select.
__global__ void __launch_bounds__(256) next_token_kernel(
    const float* __restrict__ out, int t,
    const int* __restrict__ trg, const int* __restrict__ tf_mask,
    int* __restrict__ inp)
{
    const int b = blockIdx.x;
    const float* row = out + (size_t)b * (64u * 32000u) + (size_t)t * 32000u;
    float best = -__builtin_inff();
    int   bidx = 0;
    for (int v = threadIdx.x; v < 32000; v += 256) {
        const float x = row[v];
        if (x > best) { best = x; bidx = v; }   // strictly > keeps first index
    }
    __shared__ float sv[256];
    __shared__ int   si[256];
    sv[threadIdx.x] = best; si[threadIdx.x] = bidx;
    __syncthreads();
    for (int s = 128; s > 0; s >>= 1) {
        if (threadIdx.x < (unsigned)s) {
            const float ov = sv[threadIdx.x + s];
            const int   oi = si[threadIdx.x + s];
            if (ov > sv[threadIdx.x] || (ov == sv[threadIdx.x] && oi < si[threadIdx.x])) {
                sv[threadIdx.x] = ov; si[threadIdx.x] = oi;
            }
        }
        __syncthreads();
    }
    if (threadIdx.x == 0)
        inp[b] = (tf_mask[t] > 0) ? trg[b * 64 + t] : si[0];
}

// f32 -> bf16 bulk conversion.
__global__ void __launch_bounds__(256) cvt_kernel(
    const float* __restrict__ in, unsigned short* __restrict__ o, int n)
{
    const int i = blockIdx.x * 256 + threadIdx.x;
    if (i < n) o[i] = f2bf(in[i]);
}

// Pre-gather encoder inputs: x_enc[s][b][e] = bf16(enc_emb[src[b][s]][e]).
__global__ void __launch_bounds__(256) gather_enc_kernel(
    const int* __restrict__ src, const float* __restrict__ emb,
    unsigned short* __restrict__ xo)
{
    const int id = blockIdx.x * 256 + threadIdx.x;  // exactly 128*64*512 threads
    const int e  = id & 511;
    const int b  = (id >> 9) & 63;
    const int s  = id >> 15;
    const int tok = src[b * 128 + s];
    xo[id] = f2bf(emb[(size_t)tok * 512 + e]);
}

// Decoder embedding for current input tokens.
__global__ void __launch_bounds__(256) embed_dec_kernel(
    const int* __restrict__ inp, const float* __restrict__ emb,
    unsigned short* __restrict__ xo)
{
    const int b   = blockIdx.x;
    const int tok = inp[b];
    for (int e = threadIdx.x; e < 512; e += 256)
        xo[b * 512 + e] = f2bf(emb[(size_t)tok * 512 + e]);
}

// Zero h0, c0; seed decoder input with trg[:,0].
__global__ void __launch_bounds__(256) init_state_kernel(
    unsigned short* __restrict__ h0, float* __restrict__ c0,
    int* __restrict__ inp, const int* __restrict__ trg)
{
    const int id = blockIdx.x * 256 + threadIdx.x;  // exactly 64*1024 threads
    h0[id] = 0;
    c0[id] = 0.0f;
    if (id < 64) inp[id] = trg[id * 64];
}

// out[:, 0, :] = 0
__global__ void __launch_bounds__(256) zero_t0_kernel(float* __restrict__ out)
{
    const int id = blockIdx.x * 256 + threadIdx.x;  // exactly 64*32000 threads
    const int b  = id / 32000;
    const int v  = id % 32000;
    out[(size_t)b * (64u * 32000u) + (size_t)v] = 0.0f;
}

// ---------------------------------------------------------------------------

extern "C" void kernel_launch(void* const* d_in, const int* in_sizes, int n_in,
                              void* d_out, int out_size, void* d_ws, size_t ws_size,
                              hipStream_t stream) {
    (void)in_sizes; (void)n_in; (void)out_size; (void)ws_size;

    const int*   src     = (const int*)  d_in[0];
    const int*   trg     = (const int*)  d_in[1];
    const int*   tf_mask = (const int*)  d_in[2];
    const float* enc_emb = (const float*)d_in[3];
    const float* dec_emb = (const float*)d_in[4];
    const float* Wih_e   = (const float*)d_in[5];
    const float* Whh_e   = (const float*)d_in[6];
    const float* b_e     = (const float*)d_in[7];
    const float* Wih_d   = (const float*)d_in[8];
    const float* Whh_d   = (const float*)d_in[9];
    const float* b_d     = (const float*)d_in[10];
    const float* Wfc     = (const float*)d_in[11];
    const float* bfc     = (const float*)d_in[12];
    float* out = (float*)d_out;

    // ---- carve workspace (256B-aligned) ----
    char*  ws  = (char*)d_ws;
    size_t off = 0;
    auto carve = [&](size_t bytes) -> char* {
        char* p = ws + off;
        off = (off + bytes + 255) & ~(size_t)255;
        return p;
    };
    unsigned short* wihe = (unsigned short*)carve((size_t)4096 * 512  * 2);
    unsigned short* whhe = (unsigned short*)carve((size_t)4096 * 1024 * 2);
    unsigned short* wihd = (unsigned short*)carve((size_t)4096 * 512  * 2);
    unsigned short* whhd = (unsigned short*)carve((size_t)4096 * 1024 * 2);
    unsigned short* wfcb = (unsigned short*)carve((size_t)32000 * 1024 * 2);
    unsigned short* xenc = (unsigned short*)carve((size_t)128 * 64 * 512 * 2);
    unsigned short* xdec = (unsigned short*)carve((size_t)64 * 512 * 2);
    unsigned short* hA   = (unsigned short*)carve((size_t)64 * 1024 * 2);
    unsigned short* hB   = (unsigned short*)carve((size_t)64 * 1024 * 2);
    float*          cst  = (float*)         carve((size_t)64 * 1024 * 4);
    int*            inp  = (int*)           carve((size_t)64 * 4);

    // ---- one-time prep (graph-safe, deterministic) ----
    cvt_kernel<<<(4096 * 512  + 255) / 256, 256, 0, stream>>>(Wih_e, wihe, 4096 * 512);
    cvt_kernel<<<(4096 * 1024 + 255) / 256, 256, 0, stream>>>(Whh_e, whhe, 4096 * 1024);
    cvt_kernel<<<(4096 * 512  + 255) / 256, 256, 0, stream>>>(Wih_d, wihd, 4096 * 512);
    cvt_kernel<<<(4096 * 1024 + 255) / 256, 256, 0, stream>>>(Whh_d, whhd, 4096 * 1024);
    cvt_kernel<<<(32000 * 1024 + 255) / 256, 256, 0, stream>>>(Wfc, wfcb, 32000 * 1024);

    gather_enc_kernel<<<(128 * 64 * 512) / 256, 256, 0, stream>>>(src, enc_emb, xenc);
    init_state_kernel<<<(64 * 1024) / 256, 256, 0, stream>>>(hA, cst, inp, trg);
    zero_t0_kernel<<<(64 * 32000) / 256, 256, 0, stream>>>(out);

    unsigned short* hcur = hA;
    unsigned short* hnxt = hB;

    // ---- encoder scan: 128 sequential LSTM steps ----
    for (int s = 0; s < 128; ++s) {
        lstm_step_kernel<<<64, 256, 0, stream>>>(
            xenc + (size_t)s * 64 * 512, hcur, wihe, whhe, b_e, cst, hnxt);
        unsigned short* tmp = hcur; hcur = hnxt; hnxt = tmp;
    }

    // ---- decoder scan: t = 1..63 ----
    for (int t = 1; t < 64; ++t) {
        embed_dec_kernel<<<64, 256, 0, stream>>>(inp, dec_emb, xdec);
        lstm_step_kernel<<<64, 256, 0, stream>>>(
            xdec, hcur, wihd, whhd, b_d, cst, hnxt);
        unsigned short* tmp = hcur; hcur = hnxt; hnxt = tmp;
        logits_kernel<<<1000, 256, 0, stream>>>(hcur, wfcb, bfc, out, t);
        next_token_kernel<<<64, 256, 0, stream>>>(out, t, trg, tf_mask, inp);
    }
}
